// HEATNet4_32890859553603
// MI455X (gfx1250) — matrix operand order repaired and checked by reading, not compiled, not run on hardware
//
#include <hip/hip_runtime.h>

// ---------------------------------------------------------------------------
// HEATNet forward on MI455X (gfx1250).  All dense GEMMs run through
// v_wmma_f32_16x16x32_bf16 (bf16 inputs, f32 accumulate).  Each wave computes
// a 16x64 output strip (4 accumulators) so the A fragment is reused across 4
// B tiles: 4 WMMAs per 10 b128 loads.  Edge softmax aggregation uses atomics
// (segment max via monotonic-uint atomicMax).
// ---------------------------------------------------------------------------

typedef __attribute__((ext_vector_type(16))) __bf16 v16bf;
typedef __attribute__((ext_vector_type(8)))  float  v8f;

#define C_NI   20000
#define C_NG   10000
#define C_NT   5000
#define C_NTOT 35000
#define C_D    256
#define C_DIN  512
#define C_E    100000
#define C_H    8
#define C_DK   32
#define C_L    2

// ---------------- helpers ----------------

static __device__ __forceinline__ unsigned short f2bf(float f) {
  unsigned u = __float_as_uint(f);
  unsigned r = u + 0x7FFFu + ((u >> 16) & 1u);   // round-to-nearest-even
  return (unsigned short)(r >> 16);
}

// monotonic float<->uint mapping for atomicMax on floats
static __device__ __forceinline__ unsigned mono(float f) {
  unsigned u = __float_as_uint(f);
  return (u & 0x80000000u) ? ~u : (u | 0x80000000u);
}
static __device__ __forceinline__ float unmono(unsigned u) {
  unsigned b = (u & 0x80000000u) ? (u & 0x7FFFFFFFu) : ~u;
  return __uint_as_float(b);
}

// ---------------- elementwise kernels ----------------

__global__ void k_fill_u32(unsigned* __restrict__ p, unsigned v, long n) {
  long i = (long)blockIdx.x * blockDim.x + threadIdx.x;
  if (i < n) p[i] = v;
}

__global__ void k_cvt_bf16(const float* __restrict__ s, unsigned short* __restrict__ d,
                           long n, float scale) {
  long i = (long)blockIdx.x * blockDim.x + threadIdx.x;
  if (i < n) d[i] = f2bf(s[i] * scale);
}

// W[K][N] (row-major f32) -> Wt[N][K] (row-major bf16)
__global__ void k_cvt_transpose_bf16(const float* __restrict__ s, unsigned short* __restrict__ d,
                                     int K, int N) {
  long i = (long)blockIdx.x * blockDim.x + threadIdx.x;
  if (i < (long)K * N) {
    int r = (int)(i / N), c = (int)(i % N);
    d[(long)c * K + r] = f2bf(s[i]);
  }
}

// Xbf[i][j] = bf16(X[i][j] + g[j])   (row-broadcast add, fused with conversion)
__global__ void k_add_bc_cvt(const float* __restrict__ X, const float* __restrict__ g,
                             unsigned short* __restrict__ d, long M, int K) {
  long i = (long)blockIdx.x * blockDim.x + threadIdx.x;
  if (i < M * (long)K) d[i] = f2bf(X[i] + g[(int)(i % K)]);
}

// gated skip: h = trans*sigmoid(skip[l,t]) + h*(1-sigmoid(skip[l,t]))
__global__ void k_combine(float* __restrict__ h, const float* __restrict__ trans,
                          const float* __restrict__ skip, int l, int t, long n) {
  long i = (long)blockIdx.x * blockDim.x + threadIdx.x;
  if (i < n) {
    float a = 1.0f / (1.0f + __expf(-skip[l * 3 + t]));
    h[i] = trans[i] * a + h[i] * (1.0f - a);
  }
}

// column mean via atomics: mean[col] += h[i]/Nrows   (D = 256, power of two)
__global__ void k_mean_acc(const float* __restrict__ h, float* __restrict__ mean, long n,
                           float inv_rows) {
  long i = (long)blockIdx.x * blockDim.x + threadIdx.x;
  if (i < n) atomicAdd(&mean[(int)(i & (C_D - 1))], h[i] * inv_rows);
}

// y[n] = b[n] + sum_k x[k]*W[k*N+n]   (single-row matvec, W row-major [K,N])
__global__ void k_matvec(const float* __restrict__ x, const float* __restrict__ W,
                         const float* __restrict__ b, float* __restrict__ y, int K, int N) {
  int n = blockIdx.x * blockDim.x + threadIdx.x;
  if (n >= N) return;
  float acc = b ? b[n] : 0.0f;
  for (int k = 0; k < K; ++k) acc = fmaf(x[k], W[(long)k * N + n], acc);
  y[n] = acc;
}

// ---------------- WMMA GEMM ----------------
// C[M,N] = A[M,K] (bf16 row-major) * Bt[N,K]^T (bf16, stored N-major) + bias
// One wave per 16x64 C strip (4 N-tiles, A fragment reused); 4 waves stacked
// along M per 128-thread block.  N must be a multiple of 64 (true here).
// M may be ragged: tiles = ceil(M/16), stores row-guarded (over-read of A for
// ghost rows stays inside the workspace and is never stored).

static __device__ __forceinline__ v16bf load_frag_a(const unsigned short* __restrict__ A,
                                                    int lda, int mBase, int k0, int lane) {
  // ISA 7.12.2: 16-bit A 16x32: lane&15 = row M; lane>>4 selects K-halves
  int row = mBase + (lane & 15);
  int hk  = (lane >> 4) << 3;                        // 0 or 8
  const uint4* p0 = (const uint4*)(A + (long)row * lda + k0 + hk);
  const uint4* p1 = (const uint4*)(A + (long)row * lda + k0 + 16 + hk);
  union { uint4 u[2]; v16bf v; } f;
  f.u[0] = *p0;                                      // K = k0+hk .. +7
  f.u[1] = *p1;                                      // K = k0+16+hk .. +7
  return f.v;
}

static __device__ __forceinline__ v16bf load_frag_b(const unsigned short* __restrict__ Bt,
                                                    int ldb, int nBase, int k0, int lane) {
  // B 32x16: lane&15 = col N; lane>>4 selects K-halves (same as A)
  int col = nBase + (lane & 15);
  int kb  = k0 + ((lane >> 4) << 3);                 // +0 or +8
  const uint4* p = (const uint4*)(Bt + (long)col * ldb + kb);
  union { uint4 u[2]; v16bf v; } f;
  f.u[0] = p[0];                                     // K = kb .. kb+7
  f.u[1] = p[2];                                     // K = kb+16 .. kb+23
  return f.v;
}

__global__ void __launch_bounds__(128)
k_wmma_gemm(const unsigned short* __restrict__ A, const unsigned short* __restrict__ Bt,
            const float* __restrict__ bias, float* __restrict__ C, int M, int N, int K) {
  int lane  = threadIdx.x & 31;
  int wave  = threadIdx.x >> 5;
  int mBase = (blockIdx.y * 4 + wave) * 16;
  int nBase = blockIdx.x * 64;
  if (mBase >= M) return;                            // whole-wave uniform exit
  v8f acc[4] = {};
  #pragma unroll 2
  for (int k0 = 0; k0 < K; k0 += 32) {
    v16bf a  = load_frag_a(A, K, mBase, k0, lane);
    v16bf b0 = load_frag_b(Bt, K, nBase +  0, k0, lane);
    v16bf b1 = load_frag_b(Bt, K, nBase + 16, k0, lane);
    v16bf b2 = load_frag_b(Bt, K, nBase + 32, k0, lane);
    v16bf b3 = load_frag_b(Bt, K, nBase + 48, k0, lane);
    acc[0] = __builtin_amdgcn_wmma_f32_16x16x32_bf16(false, a, false, b0, (short)0, acc[0],
                                                     false, false);
    acc[1] = __builtin_amdgcn_wmma_f32_16x16x32_bf16(false, a, false, b1, (short)0, acc[1],
                                                     false, false);
    acc[2] = __builtin_amdgcn_wmma_f32_16x16x32_bf16(false, a, false, b2, (short)0, acc[2],
                                                     false, false);
    acc[3] = __builtin_amdgcn_wmma_f32_16x16x32_bf16(false, a, false, b3, (short)0, acc[3],
                                                     false, false);
  }
  // C layout: lane&15 = col; VGPR v -> row v + 8*(lane>>4)
  int rbase = mBase + ((lane >> 4) << 3);
  #pragma unroll
  for (int nt = 0; nt < 4; ++nt) {
    int col  = nBase + nt * 16 + (lane & 15);
    float bv = bias ? bias[col] : 0.0f;
    #pragma unroll
    for (int v = 0; v < 8; ++v) {
      int row = rbase + v;
      if (row < M) C[(long)row * N + col] = acc[nt][v] + bv;
    }
  }
}

// ---------------- edge-softmax kernels (one thread per (edge, head)) ----------------

__global__ void k_edge_score_max(const float* __restrict__ q, const float* __restrict__ k,
                                 const int* __restrict__ src, const int* __restrict__ dst,
                                 const float* __restrict__ sim, const float* __restrict__ ew,
                                 const float* __restrict__ eb, int l,
                                 float* __restrict__ score, unsigned* __restrict__ mmax) {
  int tid = blockIdx.x * blockDim.x + threadIdx.x;
  if (tid >= C_E * C_H) return;
  int e = tid >> 3, h = tid & 7;
  int s = src[e], d = dst[e];
  const float4* qp = (const float4*)(q + (long)d * C_D + h * C_DK);
  const float4* kp = (const float4*)(k + (long)s * C_D + h * C_DK);
  float acc = 0.0f;
  #pragma unroll
  for (int i = 0; i < 8; ++i) {
    float4 a = qp[i], b = kp[i];
    acc = fmaf(a.x, b.x, fmaf(a.y, b.y, fmaf(a.z, b.z, fmaf(a.w, b.w, acc))));
  }
  float ea = sim[e] * ew[l] + eb[l];
  float sc = acc * ea * 0.17677669529663687f;        // 1/sqrt(DK=32)
  score[tid] = sc;
  atomicMax(&mmax[(long)d * C_H + h], mono(sc));
}

__global__ void k_edge_exp_sum(const float* __restrict__ score, const unsigned* __restrict__ mmax,
                               const int* __restrict__ dst, float* __restrict__ ex,
                               float* __restrict__ ssum) {
  int tid = blockIdx.x * blockDim.x + threadIdx.x;
  if (tid >= C_E * C_H) return;
  int e = tid >> 3, h = tid & 7;
  int d = dst[e];
  float m = unmono(mmax[(long)d * C_H + h]);
  float v = __expf(score[tid] - m);
  ex[tid] = v;
  atomicAdd(&ssum[(long)d * C_H + h], v);
}

__global__ void k_edge_scatter(const float* __restrict__ ex, const float* __restrict__ ssum,
                               const float* __restrict__ v, const int* __restrict__ src,
                               const int* __restrict__ dst, float* __restrict__ agg) {
  int tid = blockIdx.x * blockDim.x + threadIdx.x;
  if (tid >= C_E * C_H) return;
  int e = tid >> 3, h = tid & 7;
  int s = src[e], d = dst[e];
  float attn = ex[tid] / ssum[(long)d * C_H + h];
  const float* vp = v + (long)s * C_D + h * C_DK;
  float* ap = agg + (long)d * C_D + h * C_DK;
  #pragma unroll
  for (int i = 0; i < C_DK; ++i) atomicAdd(&ap[i], vp[i] * attn);
}

// ---------------- host orchestration ----------------

extern "C" void kernel_launch(void* const* d_in, const int* in_sizes, int n_in,
                              void* d_out, int out_size, void* d_ws, size_t ws_size,
                              hipStream_t stream) {
  (void)in_sizes; (void)n_in; (void)out_size; (void)ws_size;

  const float* feat[3] = {(const float*)d_in[0], (const float*)d_in[1], (const float*)d_in[2]};
  const float* adapt_w = (const float*)d_in[3];
  const float* adapt_b = (const float*)d_in[4];
  const float* kW = (const float*)d_in[5];
  const float* kB = (const float*)d_in[6];
  const float* qW = (const float*)d_in[7];
  const float* qB = (const float*)d_in[8];
  const float* vW = (const float*)d_in[9];
  const float* vB = (const float*)d_in[10];
  const float* aW = (const float*)d_in[11];
  const float* aB = (const float*)d_in[12];
  const float* e_w = (const float*)d_in[13];
  const float* e_b = (const float*)d_in[14];
  const float* skip = (const float*)d_in[15];
  const float* pred_w = (const float*)d_in[16];
  const float* pred_b = (const float*)d_in[17];
  // d_in[18] = attn_w: mathematically unused (softmax over singleton axis == 1)
  const float* head1_w = (const float*)d_in[19];
  const float* head1_b = (const float*)d_in[20];
  const float* head_w = (const float*)d_in[21];
  const float* head_b = (const float*)d_in[22];
  const float* sims[6] = {(const float*)d_in[23], (const float*)d_in[24], (const float*)d_in[25],
                          (const float*)d_in[26], (const float*)d_in[27], (const float*)d_in[28]};
  const int* srcs[6] = {(const int*)d_in[29], (const int*)d_in[31], (const int*)d_in[33],
                        (const int*)d_in[35], (const int*)d_in[37], (const int*)d_in[39]};
  const int* dsts[6] = {(const int*)d_in[30], (const int*)d_in[32], (const int*)d_in[34],
                        (const int*)d_in[36], (const int*)d_in[38], (const int*)d_in[40]};
  const int stA[6] = {0, 1, 0, 2, 1, 2};
  const int dtA[6] = {1, 0, 2, 0, 2, 1};
  const long Ns[3]  = {C_NI, C_NG, C_NT};
  const long off[3] = {0, C_NI, C_NI + C_NG};        // row offsets into [NTOT, D] buffers

  // ---- workspace bump allocator (256B aligned) ----
  char* wp = (char*)d_ws;
  auto alloc = [&](size_t bytes) -> void* {
    void* r = (void*)wp;
    wp += (bytes + 255) & ~(size_t)255;
    return r;
  };
  float* h    = (float*)alloc((size_t)C_NTOT * C_D * 4);
  float* qb   = (float*)alloc((size_t)C_NTOT * C_D * 4);
  float* kb   = (float*)alloc((size_t)C_NTOT * C_D * 4);
  float* vb   = (float*)alloc((size_t)C_NTOT * C_D * 4);
  float* agg  = (float*)alloc((size_t)C_NTOT * C_D * 4);
  float* trans = (float*)alloc((size_t)C_NI * C_D * 4);
  unsigned short* hbf   = (unsigned short*)alloc((size_t)C_NTOT * C_D * 2);
  unsigned short* actbf = (unsigned short*)alloc((size_t)C_NI * C_DIN * 2);
  unsigned short* adaptWt = (unsigned short*)alloc((size_t)3 * C_DIN * C_D * 2);
  unsigned short* kWt = (unsigned short*)alloc((size_t)C_L * 3 * C_D * C_D * 2);
  unsigned short* qWt = (unsigned short*)alloc((size_t)C_L * 3 * C_D * C_D * 2);
  unsigned short* vWt = (unsigned short*)alloc((size_t)C_L * 3 * C_D * C_D * 2);
  unsigned short* aWt = (unsigned short*)alloc((size_t)C_L * 3 * C_D * C_D * 2);
  unsigned short* headWt = (unsigned short*)alloc((size_t)C_DIN * C_DIN * 2);
  float* score = (float*)alloc((size_t)C_E * C_H * 4);
  float* exb   = (float*)alloc((size_t)C_E * C_H * 4);
  unsigned* mbuf = (unsigned*)alloc((size_t)C_NI * C_H * 4);
  float* sbuf  = (float*)alloc((size_t)C_NI * C_H * 4);
  float* mean0 = (float*)alloc(C_D * 4);
  float* outh0 = (float*)alloc(C_D * 4);
  float* gvec  = (float*)alloc(C_DIN * 4);

  auto ce = [](long n) { return dim3((unsigned)((n + 255) / 256)); };
  auto gemm = [&](const unsigned short* A, const unsigned short* Bt, const float* bias,
                  float* C, int M, int N, int K) {
    int mtiles = (M + 15) / 16;                      // ceil: handles M=5000
    dim3 grid((unsigned)(N / 64), (unsigned)((mtiles + 3) / 4));
    k_wmma_gemm<<<grid, dim3(128), 0, stream>>>(A, Bt, bias, C, M, N, K);
  };

  // ---- weight prep: f32 -> bf16, transposed to [N][K] ----
  for (int t = 0; t < 3; ++t)
    k_cvt_transpose_bf16<<<ce((long)C_DIN * C_D), 256, 0, stream>>>(
        adapt_w + (long)t * C_DIN * C_D, adaptWt + (long)t * C_DIN * C_D, C_DIN, C_D);
  for (int i = 0; i < C_L * 3; ++i) {
    long o = (long)i * C_D * C_D;
    k_cvt_transpose_bf16<<<ce((long)C_D * C_D), 256, 0, stream>>>(kW + o, kWt + o, C_D, C_D);
    k_cvt_transpose_bf16<<<ce((long)C_D * C_D), 256, 0, stream>>>(qW + o, qWt + o, C_D, C_D);
    k_cvt_transpose_bf16<<<ce((long)C_D * C_D), 256, 0, stream>>>(vW + o, vWt + o, C_D, C_D);
    k_cvt_transpose_bf16<<<ce((long)C_D * C_D), 256, 0, stream>>>(aW + o, aWt + o, C_D, C_D);
  }
  k_cvt_transpose_bf16<<<ce((long)C_DIN * C_DIN), 256, 0, stream>>>(head_w, headWt, C_DIN, C_DIN);

  // ---- adapter: h[t] = feat[t] @ adapt_w[t] + adapt_b[t] ----
  for (int t = 0; t < 3; ++t) {
    k_cvt_bf16<<<ce(Ns[t] * C_DIN), 256, 0, stream>>>(feat[t], actbf, Ns[t] * C_DIN, 1.0f);
    gemm(actbf, adaptWt + (long)t * C_DIN * C_D, adapt_b + t * C_D,
         h + off[t] * C_D, (int)Ns[t], C_D, C_DIN);
  }

  // ---- graph layers ----
  for (int l = 0; l < C_L; ++l) {
    k_cvt_bf16<<<ce((long)C_NTOT * C_D), 256, 0, stream>>>(h, hbf, (long)C_NTOT * C_D, 1.0f);
    for (int t = 0; t < 3; ++t) {
      long wo = (long)(l * 3 + t) * C_D * C_D;
      const float* bo_k = kB + (long)(l * 3 + t) * C_D;
      const float* bo_q = qB + (long)(l * 3 + t) * C_D;
      const float* bo_v = vB + (long)(l * 3 + t) * C_D;
      gemm(hbf + off[t] * C_D, kWt + wo, bo_k, kb + off[t] * C_D, (int)Ns[t], C_D, C_D);
      gemm(hbf + off[t] * C_D, qWt + wo, bo_q, qb + off[t] * C_D, (int)Ns[t], C_D, C_D);
      gemm(hbf + off[t] * C_D, vWt + wo, bo_v, vb + off[t] * C_D, (int)Ns[t], C_D, C_D);
    }
    k_fill_u32<<<ce((long)C_NTOT * C_D), 256, 0, stream>>>((unsigned*)agg, 0u, (long)C_NTOT * C_D);
    for (int et = 0; et < 6; ++et) {
      int s = stA[et], d = dtA[et];
      long ndh = Ns[d] * C_H;
      k_fill_u32<<<ce(ndh), 256, 0, stream>>>(mbuf, 0u, ndh);            // mono(-inf)
      k_fill_u32<<<ce(ndh), 256, 0, stream>>>((unsigned*)sbuf, 0u, ndh); // 0.0f
      k_edge_score_max<<<ce((long)C_E * C_H), 256, 0, stream>>>(
          qb + off[d] * C_D, kb + off[s] * C_D, srcs[et], dsts[et], sims[et],
          e_w, e_b, l, score, mbuf);
      k_edge_exp_sum<<<ce((long)C_E * C_H), 256, 0, stream>>>(score, mbuf, dsts[et], exb, sbuf);
      k_edge_scatter<<<ce((long)C_E * C_H), 256, 0, stream>>>(
          exb, sbuf, vb + off[s] * C_D, srcs[et], dsts[et], agg + off[d] * C_D);
    }
    for (int t = 0; t < 3; ++t) {
      // trans = (agg/2) @ a_w[l,t] + a_b[l,t]  (scale fused into bf16 conversion)
      k_cvt_bf16<<<ce(Ns[t] * C_D), 256, 0, stream>>>(agg + off[t] * C_D, actbf,
                                                      Ns[t] * C_D, 0.5f);
      gemm(actbf, aWt + (long)(l * 3 + t) * C_D * C_D, aB + (long)(l * 3 + t) * C_D,
           trans, (int)Ns[t], C_D, C_D);
      k_combine<<<ce(Ns[t] * C_D), 256, 0, stream>>>(h + off[t] * C_D, trans, skip, l, t,
                                                     Ns[t] * C_D);
    }
  }

  // ---- pooling head (only type 0 reaches the output; attn softmax == 1) ----
  k_fill_u32<<<1, 256, 0, stream>>>((unsigned*)mean0, 0u, C_D);
  k_mean_acc<<<ce((long)C_NI * C_D), 256, 0, stream>>>(h, mean0, (long)C_NI * C_D,
                                                       1.0f / (float)C_NI);
  k_matvec<<<1, 256, 0, stream>>>(mean0, pred_w, pred_b, outh0, C_D, C_D);
  k_matvec<<<2, 256, 0, stream>>>(outh0, head1_w, head1_b, gvec, C_D, C_DIN);

  // ---- final: out = (feat_image + g) @ head_w + head_b ----
  k_add_bc_cvt<<<ce((long)C_NI * C_DIN), 256, 0, stream>>>(feat[0], gvec, actbf,
                                                           (long)C_NI, C_DIN);
  gemm(actbf, headWt, head_b, (float*)d_out, C_NI, C_DIN, C_DIN);
}
